// SelfAttentionLayer_46712064311858
// MI455X (gfx1250) — compile-verified, matrix operand order checked by reading, not run on hardware
//
#include <hip/hip_runtime.h>
#include <hip/hip_bf16.h>

typedef __attribute__((ext_vector_type(16))) _Float16 v16h;
typedef __attribute__((ext_vector_type(8)))  _Float16 v8h;
typedef __attribute__((ext_vector_type(8)))  float    v8f;

#define T_SEQ 4096
#define BATCH 4
#define DDIM  256

// ---- WMMA fragment loaders (CDNA5 16-bit A/B layout, wave32) ----
// A 16x32: lanes 0-15 row M=lane, halves = K {0..7,16..23}; lanes 16-31 same row, +8.
// B 32x16 mirrors A with N per lane -> loading Kt rows / Vt rows works identically.
__device__ __forceinline__ v16h load_frag_h16(const _Float16* base, int ld, int k0, int lane) {
    int r  = lane & 15;
    int kh = (lane >> 4) << 3;
    const _Float16* p = base + (size_t)r * ld + k0 + kh;
    v8h lo = *(const v8h*)(p);
    v8h hi = *(const v8h*)(p + 16);
    v16h out;
#pragma unroll
    for (int i = 0; i < 8; i++) { out[i] = lo[i]; out[i + 8] = hi[i]; }
    return out;
}

__device__ __forceinline__ v16h load_frag_f32row(const float* base, int ld, int k0, int lane) {
    int r  = lane & 15;
    int kh = (lane >> 4) << 3;
    const float* p = base + (size_t)r * ld + k0 + kh;
    v16h out;
#pragma unroll
    for (int i = 0; i < 8; i++) { out[i] = (_Float16)p[i]; out[i + 8] = (_Float16)p[i + 16]; }
    return out;
}

// B fragment from row-major W[H][D]: column n per lane, K strided by ld.
__device__ __forceinline__ v16h load_frag_w(const float* Wn, int ld, int k0, int lane) {
    int kh = (lane >> 4) << 3;
    int n  = lane & 15;
    v16h out;
#pragma unroll
    for (int p = 0; p < 8; p++) {
        int kb = k0 + ((p & 3) << 1) + ((p >> 2) << 4) + kh;
        out[2 * p]     = (_Float16)Wn[(size_t)kb * ld + n];
        out[2 * p + 1] = (_Float16)Wn[(size_t)(kb + 1) * ld + n];
    }
    return out;
}

// ---- Kernel 1: Q/K/V projections via WMMA (fp32 -> f16) ----
// Qh,Kh: [B][T][256] f16 ; Vt: [B][256][T] f16 (transposed for PV B-operand)
__global__ __launch_bounds__(256) void proj_kernel(
    const float* __restrict__ X,
    const float* __restrict__ Wq, const float* __restrict__ Wk, const float* __restrict__ Wv,
    _Float16* __restrict__ Qh, _Float16* __restrict__ Kh, _Float16* __restrict__ Vt) {
    int lane = threadIdx.x & 31;
    int wave = threadIdx.x >> 5;
    int task = blockIdx.x * 8 + wave;          // 49152 wave-tasks
    int nt = task & 15; task >>= 4;
    int wmat = task % 3;
    int mt   = task / 3;                       // 0..1023
    int m0 = mt << 4;
    int n0 = nt << 4;
    const float* W = (wmat == 0) ? Wq : ((wmat == 1) ? Wk : Wv);

    v8f acc = {};
#pragma unroll
    for (int kk = 0; kk < 8; kk++) {
        v16h a = load_frag_f32row(X + (size_t)m0 * 256, 256, kk * 32, lane);
        v16h b = load_frag_w(W + n0, 256, kk * 32, lane);
        acc = __builtin_amdgcn_wmma_f32_16x16x32_f16(false, a, false, b, (short)0, acc, false, false);
    }
    int n    = lane & 15;
    int moff = (lane >> 4) << 3;
#pragma unroll
    for (int r = 0; r < 8; r++) {
        int m = m0 + r + moff;                 // global row = t*4 + b
        int t = m >> 2, bb = m & 3;
        _Float16 val = (_Float16)acc[r];
        if (wmat == 0)      Qh[(((size_t)bb * T_SEQ + t) << 8) + n0 + n] = val;
        else if (wmat == 1) Kh[(((size_t)bb * T_SEQ + t) << 8) + n0 + n] = val;
        else                Vt[((size_t)bb * DDIM + n0 + n) * T_SEQ + t] = val;
    }
}

// ---- Kernel 2: fused causal attention, 16-row strip per WG, 256KB LDS strip ----
__global__ __launch_bounds__(256) void attn_kernel(
    const _Float16* __restrict__ Qh, const _Float16* __restrict__ Kh,
    const _Float16* __restrict__ Vt,
    float* __restrict__ out_res, float* __restrict__ out_attn) {
    __shared__ float P[16 * 4096];             // 256 KB score/prob strip
    __shared__ float red[256];
    __shared__ float rowMax[16];
    __shared__ float rowInv[16];

    int lane = threadIdx.x & 31;
    int wave = threadIdx.x >> 5;
    int wg = blockIdx.x;
    int b  = wg & 3;
    int it = wg >> 2;                          // 0..255
    int i0 = it << 4;
    int nTiles = it + 1;
    int ncols  = i0 + 16;

    // zero-pad 16 cols when K-extent isn't a multiple of 32 (read as zeros in PV)
    if (nTiles & 1)
        P[((threadIdx.x >> 4) * 4096) + ncols + (threadIdx.x & 15)] = 0.0f;

    const _Float16* Qb = Qh + (((size_t)b * T_SEQ + i0) << 8);
    const _Float16* Kb = Kh + (((size_t)b * T_SEQ) << 8);

    v16h aq[8];
#pragma unroll
    for (int kk = 0; kk < 8; kk++) aq[kk] = load_frag_h16(Qb, 256, kk * 32, lane);

    // Phase 1: S = Q Kt / 16 with causal mask, waves split j-tiles
    for (int jt = wave; jt < nTiles; jt += 8) {
        int j0 = jt << 4;
        v8f acc = {};
#pragma unroll
        for (int kk = 0; kk < 8; kk++) {
            v16h bk = load_frag_h16(Kb + ((size_t)j0 << 8), 256, kk * 32, lane);
            acc = __builtin_amdgcn_wmma_f32_16x16x32_f16(false, aq[kk], false, bk, (short)0, acc, false, false);
        }
        int n = lane & 15;
        int moff = (lane >> 4) << 3;
#pragma unroll
        for (int r = 0; r < 8; r++) {
            int m = r + moff;
            float s = acc[r] * 0.0625f;        // 1/int(sqrt(256))
            if (j0 + n > i0 + m) s = -3.0e38f; // causal mask
            P[m * 4096 + j0 + n] = s;
        }
    }
    __syncthreads();

    // Phase 2: row softmax (16 threads per row); P <- exp(s - max), keep 1/sum
    int row = threadIdx.x >> 4, sub = threadIdx.x & 15;
    float mx = -3.0e38f;
    for (int j = sub; j < ncols; j += 16) mx = fmaxf(mx, P[row * 4096 + j]);
    red[threadIdx.x] = mx;
    __syncthreads();
    if (sub == 0) {
        float m = red[row * 16];
#pragma unroll
        for (int i = 1; i < 16; i++) m = fmaxf(m, red[row * 16 + i]);
        rowMax[row] = m;
    }
    __syncthreads();
    float M = rowMax[row];
    float sum = 0.0f;
    for (int j = sub; j < ncols; j += 16) {
        float e = __expf(P[row * 4096 + j] - M);
        P[row * 4096 + j] = e;
        sum += e;
    }
    red[threadIdx.x] = sum;
    __syncthreads();
    if (sub == 0) {
        float s = 0.0f;
#pragma unroll
        for (int i = 0; i < 16; i++) s += red[row * 16 + i];
        rowInv[row] = 1.0f / s;
    }
    __syncthreads();

    // Phase 3: stream attn block out (normalized; zeros above diagonal)
    float* attnB = out_attn + ((size_t)b * T_SEQ + i0) * T_SEQ;
    for (int r = 0; r < 16; r++) {
        float inv = rowInv[r];
        for (int j = threadIdx.x; j < T_SEQ; j += 256) {
            float v = (j < ncols) ? P[r * 4096 + j] * inv : 0.0f;
            attnB[(size_t)r * T_SEQ + j] = v;
        }
    }

    // Phase 4: results = (P @ V) * 1/sum ; wave owns two 16-wide D tiles
    const _Float16* Vb = Vt + ((size_t)b * DDIM) * T_SEQ;
    int n0 = wave << 5;
    int nK32 = (ncols + 31) >> 5;
    v8f acc0 = {}, acc1 = {};
    for (int kk = 0; kk < nK32; kk++) {
        int j0 = kk << 5;
        int r  = lane & 15;
        int kh = (lane >> 4) << 3;
        const float* pp = P + r * 4096 + j0 + kh;
        v16h ap;
#pragma unroll
        for (int i = 0; i < 8; i++) { ap[i] = (_Float16)pp[i]; ap[i + 8] = (_Float16)pp[i + 16]; }
        v16h b0 = load_frag_h16(Vb + (size_t)n0 * T_SEQ, T_SEQ, j0, lane);
        acc0 = __builtin_amdgcn_wmma_f32_16x16x32_f16(false, ap, false, b0, (short)0, acc0, false, false);
        v16h b1 = load_frag_h16(Vb + (size_t)(n0 + 16) * T_SEQ, T_SEQ, j0, lane);
        acc1 = __builtin_amdgcn_wmma_f32_16x16x32_f16(false, ap, false, b1, (short)0, acc1, false, false);
    }
    int n = lane & 15;
    int moff = (lane >> 4) << 3;
#pragma unroll
    for (int r = 0; r < 8; r++) {
        int m = r + moff;
        float inv = rowInv[m];
        size_t o = ((size_t)(i0 + m) * BATCH + b) * DDIM;
        out_res[o + n0 + n]      = acc0[r] * inv;
        out_res[o + n0 + 16 + n] = acc1[r] * inv;
    }
}

extern "C" void kernel_launch(void* const* d_in, const int* in_sizes, int n_in,
                              void* d_out, int out_size, void* d_ws, size_t ws_size,
                              hipStream_t stream) {
    const float* X  = (const float*)d_in[0];
    const float* Wq = (const float*)d_in[1];
    const float* Wk = (const float*)d_in[2];
    const float* Wv = (const float*)d_in[3];
    float* out      = (float*)d_out;
    float* out_res  = out;                                      // [T,B,256]
    float* out_attn = out + (size_t)T_SEQ * BATCH * DDIM;       // [B,T,T]

    _Float16* Qh = (_Float16*)d_ws;                             // 8 MB
    _Float16* Kh = Qh + (size_t)BATCH * T_SEQ * DDIM;           // 8 MB
    _Float16* Vt = Kh + (size_t)BATCH * T_SEQ * DDIM;           // 8 MB

    proj_kernel<<<6144, 256, 0, stream>>>(X, Wq, Wk, Wv, Qh, Kh, Vt);
    attn_kernel<<<1024, 256, 0, stream>>>(Qh, Kh, Vt, out_res, out_attn);
}
